// MSACol_20194936225869
// MI455X (gfx1250) — compile-verified, matrix operand order checked by reading, not run on hardware
//
#include <hip/hip_runtime.h>

// ---------------------------------------------------------------------------
// MSA column attention, fused for MI455X (gfx1250, wave32, WMMA bf16).
//
//   0) k_pack  : {Wq,Wk,Wv,Wg,Wo} fp32 -> bf16, pre-swizzled into exact
//                WMMA B-fragment order (2x b128 loads per fragment later).
//   1) k_ln    : LayerNorm(m) -> x_bf16
//   2) k_proj  : x @ {Wq,Wk,Wv,Wg} -> q,k,v bf16; g = sigmoid(.+bg)
//   3) k_attn  : per (l,h): softmax(q k^T / sqrt(C)) v * g   (fused, TDM-staged)
//   4) k_out   : o @ Wo + bo -> out fp32
//
// All B-fragment streams are double-buffered in registers so WMMA overlaps
// the (L2-resident) loads instead of serializing on s_wait_loadcnt 0.
// ---------------------------------------------------------------------------

typedef __bf16 bf16x16 __attribute__((ext_vector_type(16)));
typedef float  f32x8   __attribute__((ext_vector_type(8)));
typedef unsigned int u32x4 __attribute__((ext_vector_type(4)));
typedef int    i32x4  __attribute__((ext_vector_type(4)));
typedef int    i32x8  __attribute__((ext_vector_type(8)));

union BF16x16 {
    bf16x16        v;
    unsigned short u[16];
    u32x4          q[2];
};

static constexpr int kN    = 256;          // N_SEQ
static constexpr int kL    = 384;          // L_RES
static constexpr int kD    = 256;          // M_DIM == N_HEAD*C
static constexpr int kH    = 8;            // N_HEAD
static constexpr int kC    = 32;           // C (head dim)
static constexpr int kRows = kN * kL;      // 98304 flattened (s,l) rows
static constexpr int kWElems = kD * kD;    // 65536 elements per weight
static constexpr float kScale = 0.17677669529663687f; // 1/sqrt(32)

__device__ __forceinline__ unsigned short f2bf(float f) {
    union { float f; unsigned u; } x;
    x.f = f;
    unsigned r = x.u + 0x7FFFu + ((x.u >> 16) & 1u);   // round-to-nearest-even
    return (unsigned short)(r >> 16);
}

// ---------------------------------------------------------------------------
// TDM: async 2D tile load (Global -> LDS) via the Tensor Data Mover.
// ---------------------------------------------------------------------------
__device__ __forceinline__ void tdm_load_tile_bf16(const void* gsrc,
                                                   unsigned lds_off,
                                                   unsigned tile_x,
                                                   unsigned tile_y,
                                                   unsigned stride_elems) {
    unsigned long long ga = (unsigned long long)gsrc;
    u32x4 g0;
    g0[0] = 1u;                                   // count=1, user descriptor
    g0[1] = lds_off;                              // lds_addr (bytes)
    g0[2] = (unsigned)(ga & 0xFFFFFFFFu);         // global_addr[31:0]
    g0[3] = (unsigned)((ga >> 32) & 0x01FFFFFFu)  // global_addr[56:32]
            | (2u << 30);                         // type = 2 ("image")
    i32x8 g1;
    g1[0] = (int)(1u << 16);                      // wg_mask=0, data_size=1 (2B)
    g1[1] = (int)(tile_x << 16);                  // tensor_dim0[15:0]
    g1[2] = (int)(tile_y << 16);                  // tensor_dim0 hi | dim1 lo
    g1[3] = (int)(tile_x << 16);                  // dim1 hi | tile_dim0
    g1[4] = (int)(tile_y & 0xFFFFu);              // tile_dim1 | tile_dim2=0
    g1[5] = (int)stride_elems;                    // tensor_dim0_stride[31:0]
    g1[6] = 0;
    g1[7] = 0;
    i32x4 z4 = (i32x4)(0);
#if defined(__clang_major__) && (__clang_major__ >= 23)
    i32x8 z8 = (i32x8)(0);
    __builtin_amdgcn_tensor_load_to_lds(g0, g1, z4, z4, z8, 0);
#else
    __builtin_amdgcn_tensor_load_to_lds(g0, g1, z4, z4, 0);
#endif
}

// ---------------------------------------------------------------------------
// Kernel 0: pack weights fp32 -> bf16 in B-fragment-major order.
// ---------------------------------------------------------------------------
__global__ void k_pack(const float* __restrict__ Wq, const float* __restrict__ Wk,
                       const float* __restrict__ Wv, const float* __restrict__ Wg,
                       const float* __restrict__ Wo,
                       unsigned short* __restrict__ wpk) {
    const int lane = threadIdx.x & 31;
    const int wave = threadIdx.x >> 5;
    const int half = lane >> 4;
    const int r    = lane & 15;
    const int slot = blockIdx.x * 8 + wave;       // 0..639
    const int w    = slot >> 7;
    const int rem  = slot & 127;
    const int kk   = rem >> 4;
    const int nt   = rem & 15;

    const float* Ws[5] = {Wq, Wk, Wv, Wg, Wo};
    const float* W = Ws[w];
    const int n  = nt * 16 + r;
    const int kb = kk * 32 + half * 8;

    BF16x16 b;
#pragma unroll
    for (int i = 0; i < 8; ++i) {
        b.u[i]     = f2bf(W[(size_t)(kb + i) * kD + n]);
        b.u[8 + i] = f2bf(W[(size_t)(kb + 16 + i) * kD + n]);
    }
    unsigned short* dst = wpk + (size_t)w * kWElems + ((rem * 32) + lane) * 16;
    u32x4* dq = (u32x4*)dst;
    dq[0] = b.q[0];
    dq[1] = b.q[1];
}

// ---------------------------------------------------------------------------
// Kernel 1: LayerNorm + bf16 convert. One wave per 256-element row.
// ---------------------------------------------------------------------------
__global__ void k_ln(const float* __restrict__ m,
                     const float* __restrict__ gamma,
                     const float* __restrict__ beta,
                     unsigned short* __restrict__ xbf) {
    const int lane = threadIdx.x & 31;
    const int wave = threadIdx.x >> 5;
    const size_t row = (size_t)blockIdx.x * 8 + wave;
    const float* x = m + row * kD;

    float v[8];
    float s = 0.f, sq = 0.f;
#pragma unroll
    for (int i = 0; i < 8; ++i) {
        float t = x[i * 32 + lane];
        v[i] = t; s += t; sq += t * t;
    }
#pragma unroll
    for (int off = 16; off > 0; off >>= 1) {
        s  += __shfl_xor(s,  off, 32);
        sq += __shfl_xor(sq, off, 32);
    }
    const float mu  = s * (1.f / kD);
    const float var = sq * (1.f / kD) - mu * mu;
    const float rs  = rsqrtf(var + 1e-5f);

    unsigned short* o = xbf + row * kD;
#pragma unroll
    for (int i = 0; i < 8; ++i) {
        int c = i * 32 + lane;
        o[c] = f2bf((v[i] - mu) * rs * gamma[c] + beta[c]);
    }
}

// ---------------------------------------------------------------------------
// GEMM inner tile: K=256 in 8 WMMA steps, B-fragments double-buffered.
// ---------------------------------------------------------------------------
__device__ __forceinline__ f32x8 gemm_tile(const BF16x16 (&a)[8],
                                           const unsigned short* __restrict__ Wp,
                                           int nt, int lane) {
    f32x8 acc = (f32x8)(0.0f);
    BF16x16 b[2];
    {
        const u32x4* fp = (const u32x4*)(Wp + (((0 * 16 + nt) * 32) + lane) * 16);
        b[0].q[0] = fp[0];
        b[0].q[1] = fp[1];
    }
#pragma unroll
    for (int kk = 0; kk < 8; ++kk) {
        if (kk < 7) {
            const u32x4* fp =
                (const u32x4*)(Wp + ((((kk + 1) * 16 + nt) * 32) + lane) * 16);
            b[(kk + 1) & 1].q[0] = fp[0];
            b[(kk + 1) & 1].q[1] = fp[1];
        }
        acc = __builtin_amdgcn_wmma_f32_16x16x32_bf16(
            false, a[kk].v, false, b[kk & 1].v, (short)0, acc, false, false);
    }
    return acc;
}

// ---------------------------------------------------------------------------
// Kernel 2: fused projections. Block = 8 waves, each wave owns 16 rows.
// ---------------------------------------------------------------------------
__global__ void k_proj(const unsigned short* __restrict__ xbf,
                       const unsigned short* __restrict__ wpk,
                       const float* __restrict__ bg,
                       unsigned short* __restrict__ qbf,
                       unsigned short* __restrict__ kbf,
                       unsigned short* __restrict__ vbf,
                       float* __restrict__ gout) {
    const int lane = threadIdx.x & 31;
    const int wave = threadIdx.x >> 5;
    const int half = lane >> 4;
    const int r    = lane & 15;
    const size_t m0 = (size_t)blockIdx.x * 128 + wave * 16;

    BF16x16 a[8];
    const unsigned short* arow = xbf + (m0 + r) * kD;
#pragma unroll
    for (int kk = 0; kk < 8; ++kk) {
        int c0 = kk * 32 + half * 8;
        a[kk].q[0] = *(const u32x4*)(arow + c0);
        a[kk].q[1] = *(const u32x4*)(arow + c0 + 16);
    }
    __builtin_prefetch(arow + 16 * kD, 0, 0);   // global_prefetch_b8

    // q, k, v projections (branch-free epilogue per weight).
    unsigned short* dsts[3] = {qbf, kbf, vbf};
    for (int wsel = 0; wsel < 3; ++wsel) {
        const unsigned short* Wp = wpk + (size_t)wsel * kWElems;
        unsigned short* dst = dsts[wsel];
        for (int nt = 0; nt < 16; ++nt) {
            const int n = nt * 16 + r;
            f32x8 acc = gemm_tile(a, Wp, nt, lane);
#pragma unroll
            for (int j = 0; j < 8; ++j) {
                size_t mr = m0 + j + half * 8;
                dst[mr * kD + n] = f2bf(acc[j]);
            }
        }
    }
    // gate projection + sigmoid.
    {
        const unsigned short* Wp = wpk + (size_t)3 * kWElems;
        for (int nt = 0; nt < 16; ++nt) {
            const int n = nt * 16 + r;
            f32x8 acc = gemm_tile(a, Wp, nt, lane);
            const float bias = bg[n];
#pragma unroll
            for (int j = 0; j < 8; ++j) {
                size_t mr = m0 + j + half * 8;
                float sv = acc[j] + bias;
                gout[mr * kD + n] = 1.0f / (1.0f + __expf(-sv));
            }
        }
    }
}

// ---------------------------------------------------------------------------
// Kernel 3: fused attention per (l,h). TDM-staged K/V, register-resident
// score matrix, half-wave shfl_xor softmax, double-buffered ds fragments.
// ---------------------------------------------------------------------------
__global__ void k_attn(const unsigned short* __restrict__ qbf,
                       const unsigned short* __restrict__ kbf,
                       const unsigned short* __restrict__ vbf,
                       const float* __restrict__ g,
                       unsigned short* __restrict__ obf) {
    __shared__ unsigned short sK [kN * kC];       // 16 KB, row-major [s][c]
    __shared__ unsigned short sVr[kN * kC];       // 16 KB, row-major [t][c]
    __shared__ unsigned short sVp[kN * kC];       // 16 KB, fragment-major
    __shared__ unsigned short sP [8 * 16 * kC];   //  8 KB (1 KB per wave)

    const int lane = threadIdx.x & 31;
    const int wave = threadIdx.x >> 5;
    const int half = lane >> 4;
    const int r    = lane & 15;
    const int l    = blockIdx.x >> 3;
    const int h    = blockIdx.x & 7;

    const size_t tile_base = (size_t)l * kD + h * kC;

    // --- TDM stage of K and V: 256 x 32 bf16 tile, row stride L*D elems ---
    if (wave == 0) {
        tdm_load_tile_bf16(kbf + tile_base, (unsigned)(size_t)(void*)sK,
                           kC, kN, kL * kD);
        tdm_load_tile_bf16(vbf + tile_base, (unsigned)(size_t)(void*)sVr,
                           kC, kN, kL * kD);
        __builtin_amdgcn_s_wait_tensorcnt(0);
    }
    __syncthreads();

    // --- One-time swizzle: V row-major -> B-fragment-major ---
    {
#pragma unroll
        for (int ss = 0; ss < 2; ++ss) {
            const int slot = threadIdx.x * 2 + ss;      // 0..511
            const int fi   = slot >> 5;                 // frag = kt*2 + ntile
            const int ln   = slot & 31;
            const int fh   = ln >> 4;
            const int fr   = ln & 15;
            const int kt   = fi >> 1;
            const int ntl  = fi & 1;
            const int kb   = kt * 32 + fh * 8;
            const int c    = ntl * 16 + fr;
            BF16x16 t;
#pragma unroll
            for (int i = 0; i < 8; ++i) {
                t.u[i]     = sVr[(kb + i)      * kC + c];
                t.u[8 + i] = sVr[(kb + 16 + i) * kC + c];
            }
            u32x4* dq = (u32x4*)(sVp + slot * 16);
            dq[0] = t.q[0];
            dq[1] = t.q[1];
        }
    }
    __syncthreads();

    unsigned short* sPw = sP + wave * 16 * kC;
    const f32x8 zero = (f32x8)(0.0f);

    for (int sb = 0; sb < 2; ++sb) {
        const int s0 = wave * 32 + sb * 16;

        BF16x16 aQ;
        {
            const unsigned short* qrow =
                qbf + ((size_t)(s0 + r) * kL + l) * kD + h * kC + half * 8;
            aQ.q[0] = *(const u32x4*)qrow;
            aQ.q[1] = *(const u32x4*)(qrow + 16);
        }

        // S = q k^T : contiguous b128 ds loads, double-buffered.
        f32x8 S[16];
        BF16x16 bK[2];
        {
            const unsigned short* kp = sK + r * kC + half * 8;
            bK[0].q[0] = *(const u32x4*)kp;
            bK[0].q[1] = *(const u32x4*)(kp + 16);
        }
#pragma unroll
        for (int t = 0; t < 16; ++t) {
            if (t < 15) {
                const unsigned short* kp = sK + ((t + 1) * 16 + r) * kC + half * 8;
                bK[(t + 1) & 1].q[0] = *(const u32x4*)kp;
                bK[(t + 1) & 1].q[1] = *(const u32x4*)(kp + 16);
            }
            S[t] = __builtin_amdgcn_wmma_f32_16x16x32_bf16(
                false, aQ.v, false, bK[t & 1].v, (short)0, zero, false, false);
        }

        // Softmax over t (rows split across the two 16-lane halves).
        float inv[8];
#pragma unroll
        for (int j = 0; j < 8; ++j) {
            float mx = -3.0e38f;
#pragma unroll
            for (int t = 0; t < 16; ++t) {
                float x = S[t][j] * kScale; S[t][j] = x; mx = fmaxf(mx, x);
            }
#pragma unroll
            for (int off = 1; off < 16; off <<= 1)
                mx = fmaxf(mx, __shfl_xor(mx, off, 32));
            float sum = 0.f;
#pragma unroll
            for (int t = 0; t < 16; ++t) {
                float e = __expf(S[t][j] - mx); S[t][j] = e; sum += e;
            }
#pragma unroll
            for (int off = 1; off < 16; off <<= 1)
                sum += __shfl_xor(sum, off, 32);
            inv[j] = 1.0f / sum;
        }

        // O = P v : P streamed through 1KB/wave LDS to A layout; V fragments
        // double-buffered one kt ahead (independent of the P stores).
        f32x8 o0 = zero, o1 = zero;
        BF16x16 b0[2], b1[2];
        {
            const u32x4* v0 = (const u32x4*)(sVp + (0 * 32 + lane) * 16);
            const u32x4* v1 = (const u32x4*)(sVp + (1 * 32 + lane) * 16);
            b0[0].q[0] = v0[0]; b0[0].q[1] = v0[1];
            b1[0].q[0] = v1[0]; b1[0].q[1] = v1[1];
        }
        for (int kt = 0; kt < 8; ++kt) {
            if (kt < 7) {
                const u32x4* v0 =
                    (const u32x4*)(sVp + (((kt + 1) * 2 + 0) * 32 + lane) * 16);
                const u32x4* v1 =
                    (const u32x4*)(sVp + (((kt + 1) * 2 + 1) * 32 + lane) * 16);
                b0[(kt + 1) & 1].q[0] = v0[0]; b0[(kt + 1) & 1].q[1] = v0[1];
                b1[(kt + 1) & 1].q[0] = v1[0]; b1[(kt + 1) & 1].q[1] = v1[1];
            }
#pragma unroll
            for (int tt = 0; tt < 2; ++tt) {
                const int t = kt * 2 + tt;
#pragma unroll
                for (int j = 0; j < 8; ++j) {
                    const int prow = j + half * 8;
                    sPw[prow * kC + tt * 16 + r] = f2bf(S[t][j] * inv[j]);
                }
            }
            BF16x16 aP;
            const unsigned short* pp = sPw + r * kC + half * 8;
            aP.q[0] = *(const u32x4*)pp;
            aP.q[1] = *(const u32x4*)(pp + 16);

            o0 = __builtin_amdgcn_wmma_f32_16x16x32_bf16(
                false, aP.v, false, b0[kt & 1].v, (short)0, o0, false, false);
            o1 = __builtin_amdgcn_wmma_f32_16x16x32_bf16(
                false, aP.v, false, b1[kt & 1].v, (short)0, o1, false, false);
        }

        // Gate and store bf16.
#pragma unroll
        for (int j = 0; j < 8; ++j) {
            const size_t s    = (size_t)s0 + j + half * 8;
            const size_t base = (s * kL + l) * kD + h * kC;
            const float g0 = g[base + r];
            const float g1 = g[base + 16 + r];
            obf[base + r]      = f2bf(o0[j] * g0);
            obf[base + 16 + r] = f2bf(o1[j] * g1);
        }
    }
}

// ---------------------------------------------------------------------------
// Kernel 4: output GEMM + bias, fp32 result (packed Wo fragments).
// ---------------------------------------------------------------------------
__global__ void k_out(const unsigned short* __restrict__ obf,
                      const unsigned short* __restrict__ wpk,
                      const float* __restrict__ bo,
                      float* __restrict__ out) {
    const int lane = threadIdx.x & 31;
    const int wave = threadIdx.x >> 5;
    const int half = lane >> 4;
    const int r    = lane & 15;
    const size_t m0 = (size_t)blockIdx.x * 128 + wave * 16;

    BF16x16 a[8];
    const unsigned short* arow = obf + (m0 + r) * kD;
#pragma unroll
    for (int kk = 0; kk < 8; ++kk) {
        int c0 = kk * 32 + half * 8;
        a[kk].q[0] = *(const u32x4*)(arow + c0);
        a[kk].q[1] = *(const u32x4*)(arow + c0 + 16);
    }

    const unsigned short* Wp = wpk + (size_t)4 * kWElems;
    for (int nt = 0; nt < 16; ++nt) {
        const int n = nt * 16 + r;
        f32x8 acc = gemm_tile(a, Wp, nt, lane);
        const float bias = bo[n];
#pragma unroll
        for (int j = 0; j < 8; ++j) {
            size_t mr = m0 + j + half * 8;
            out[mr * kD + n] = acc[j] + bias;
        }
    }
}

// ---------------------------------------------------------------------------
// Host launcher. Workspace (~337 MB):
//   x|q|k|v bf16 (48MB ea) | g fp32 (96MB) | o bf16 (48MB) | packed W (640KB)
// ---------------------------------------------------------------------------
extern "C" void kernel_launch(void* const* d_in, const int* in_sizes, int n_in,
                              void* d_out, int out_size, void* d_ws, size_t ws_size,
                              hipStream_t stream) {
    const float* m     = (const float*)d_in[0];
    const float* gamma = (const float*)d_in[1];
    const float* beta  = (const float*)d_in[2];
    const float* Wq    = (const float*)d_in[3];
    const float* Wk    = (const float*)d_in[4];
    const float* Wv    = (const float*)d_in[5];
    const float* Wg    = (const float*)d_in[6];
    const float* bg    = (const float*)d_in[7];
    const float* Wo    = (const float*)d_in[8];
    const float* bo    = (const float*)d_in[9];
    float* out = (float*)d_out;

    const size_t nElem = (size_t)kRows * kD;   // 25,165,824 elements
    unsigned short* xbf = (unsigned short*)d_ws;
    unsigned short* qbf = xbf + nElem;
    unsigned short* kbf = qbf + nElem;
    unsigned short* vbf = kbf + nElem;
    float*          gws = (float*)(vbf + nElem);
    unsigned short* obf = (unsigned short*)(gws + nElem);
    unsigned short* wpk = obf + nElem;

    k_pack<<<80,          256, 0, stream>>>(Wq, Wk, Wv, Wg, Wo, wpk);
    k_ln  <<<kRows / 8,   256, 0, stream>>>(m, gamma, beta, xbf);
    k_proj<<<kRows / 128, 256, 0, stream>>>(xbf, wpk, bg, qbf, kbf, vbf, gws);
    k_attn<<<kL * kH,     256, 0, stream>>>(qbf, kbf, vbf, gws, obf);
    k_out <<<kRows / 128, 256, 0, stream>>>(obf, wpk, bo, out);
}
